// SelfAttention_38783554683222
// MI455X (gfx1250) — compile-verified, matrix operand order checked by reading, not run on hardware
//
#include <hip/hip_runtime.h>

// ---------------------------------------------------------------------------
// CDNA5 (gfx1250) self-attention forward.
//  - f16 WMMA (v_wmma_f32_16x16x32_f16) with f32 accumulation
//  - flash-attention streaming softmax (S never touches HBM)
//  - all weight/V transposes hoisted out of hot loops -> every LDS stage is a
//    contiguous 16B copy, routed through GLOBAL_LOAD_ASYNC_TO_LDS_B128
//  - double-buffered async software pipeline in the GEMMs
// ---------------------------------------------------------------------------

typedef __attribute__((ext_vector_type(16))) _Float16 v16h;
typedef __attribute__((ext_vector_type(8)))  _Float16 v8h;
typedef __attribute__((ext_vector_type(8)))  float    v8f;
typedef __attribute__((ext_vector_type(4)))  int      v4i;

#if defined(__has_builtin)
#if __has_builtin(__builtin_amdgcn_global_load_async_to_lds_b128)
#define HAVE_ASYNC 1
#endif
#endif
#ifndef HAVE_ASYNC
#define HAVE_ASYNC 0
#endif

#if HAVE_ASYNC
typedef __attribute__((address_space(1))) v4i gv4i;
typedef __attribute__((address_space(3))) v4i lv4i;
#define ASYNC_WAIT_ALL() asm volatile("s_wait_asynccnt 0x0" ::: "memory")
#else
#define ASYNC_WAIT_ALL()
#endif

// copy 16 bytes global -> LDS (async if toolchain exposes the builtin)
__device__ inline void async_cp16(void* l, const void* g) {
#if HAVE_ASYNC
  __builtin_amdgcn_global_load_async_to_lds_b128((gv4i*)(void*)g, (lv4i*)l, 0, 0);
#else
  *(v8h*)l = *(const v8h*)g;
#endif
}

// Load a 16x32 f16 A-fragment row (or mirrored B-fragment column) from an LDS
// row pointer. Per ISA 7.12.2 (16-bit A 16x32):
//   lanes 0-15 : elems 0-7 -> K=0..7,   elems 8-15 -> K=16..23
//   lanes 16-31: elems 0-7 -> K=8..15,  elems 8-15 -> K=24..31
__device__ inline v16h frag_ld(const _Float16* rowp, int half) {
  v8h lo = *(const v8h*)(rowp + half * 8);
  v8h hi = *(const v8h*)(rowp + 16 + half * 8);
  v16h r;
#pragma unroll
  for (int i = 0; i < 8; ++i) { r[i] = lo[i]; r[i + 8] = hi[i]; }
  return r;
}

__device__ inline v8f wmma16(v16h a, v16h b, v8f c) {
  return __builtin_amdgcn_wmma_f32_16x16x32_f16(false, a, false, b,
                                                (short)0, c, false, false);
}

// ---------------------------------------------------------------------------
// f32 -> f16 elementwise (n multiple of 4)
// ---------------------------------------------------------------------------
__global__ __launch_bounds__(256) void cvt_f32_f16(const float* __restrict__ in,
                                                   _Float16* __restrict__ out,
                                                   int n) {
  int i = (blockIdx.x * blockDim.x + threadIdx.x) * 4;
  if (i < n) {
    float4 v = *(const float4*)(in + i);
    out[i + 0] = (_Float16)v.x;
    out[i + 1] = (_Float16)v.y;
    out[i + 2] = (_Float16)v.z;
    out[i + 3] = (_Float16)v.w;
  }
}

// ---------------------------------------------------------------------------
// f32 [K][N] -> f16 transposed [N][K] via 32x32 LDS tiles (one-time pass)
// ---------------------------------------------------------------------------
struct alignas(8) h4 { _Float16 v[4]; };

__global__ __launch_bounds__(256) void cvtT(const float* __restrict__ in,
                                            _Float16* __restrict__ out,
                                            int N, int K) {
  __shared__ _Float16 t[32][36];
  const int k0 = blockIdx.y * 32, n0 = blockIdx.x * 32;
  const int r = threadIdx.x >> 3;        // 0..31
  const int c = (threadIdx.x & 7) * 4;   // 0..28
  float4 v = *(const float4*)(in + (long)(k0 + r) * N + n0 + c);
  t[c + 0][r] = (_Float16)v.x;
  t[c + 1][r] = (_Float16)v.y;
  t[c + 2][r] = (_Float16)v.z;
  t[c + 3][r] = (_Float16)v.w;
  __syncthreads();
  h4 o;
#pragma unroll
  for (int i = 0; i < 4; ++i) o.v[i] = t[r][c + i];
  *(h4*)(out + (long)(n0 + r) * K + k0 + c) = o;
}

// ---------------------------------------------------------------------------
// V repack: qkv f16 [b,t,3072] (V at col 2048+h*64) -> vt [b*16+h][dh][t]
// ---------------------------------------------------------------------------
__global__ __launch_bounds__(256) void vtrans(const _Float16* __restrict__ qkv,
                                              _Float16* __restrict__ vt) {
  __shared__ _Float16 t[64][72];
  const int t0 = blockIdx.x * 64;
  const int bh = blockIdx.y;
  const int b = bh >> 4, h = bh & 15;
  const int tid = threadIdx.x;
  {
    int row = tid >> 2, col = (tid & 3) * 16;
    const _Float16* src =
        qkv + ((long)(b * 2048 + t0 + row)) * 3072 + 2048 + h * 64 + col;
    *(v8h*)&t[row][col] = *(const v8h*)(src);
    *(v8h*)&t[row][col + 8] = *(const v8h*)(src + 8);
  }
  __syncthreads();
  {
    int d = tid >> 2, ts = (tid & 3) * 16;
    v8h o0, o1;
#pragma unroll
    for (int i = 0; i < 8; ++i) { o0[i] = t[ts + i][d]; o1[i] = t[ts + 8 + i][d]; }
    _Float16* dst = vt + ((long)(bh * 64 + d)) * 2048 + t0 + ts;
    *(v8h*)dst = o0;
    *(v8h*)(dst + 8) = o1;
  }
}

// ---------------------------------------------------------------------------
// 128x128-tile GEMM: C[M,N] = A[M,K] * Bt[N,K]^T  (A, Bt row-major f16)
// 8 waves; wave w -> rows (w>>1)*32..+31 (2 subtiles), cols (w&1)*64..+63 (4).
// Double-buffered async LDS pipeline. M,N multiples of 128; K multiple of 32.
// ---------------------------------------------------------------------------
template <bool F16OUT>
__global__ __launch_bounds__(256) void gemm128(const _Float16* __restrict__ A, int lda,
                                               const _Float16* __restrict__ Bt, int ldb,
                                               void* __restrict__ Cout, int ldc,
                                               const float* __restrict__ bias, int K) {
  __shared__ _Float16 sA[2][128][40];
  __shared__ _Float16 sB[2][128][40];
  const int m0 = blockIdx.y * 128, n0 = blockIdx.x * 128;
  const int tid = threadIdx.x;
  const int w = tid >> 5, lane = tid & 31;
  const int half = lane >> 4, lanelo = lane & 15;
  const int wm = (w >> 1) * 32;   // 0,32,64,96
  const int wn = (w & 1) * 64;    // 0,64

  v8f acc[2][4];
#pragma unroll
  for (int i = 0; i < 2; ++i)
#pragma unroll
    for (int j = 0; j < 4; ++j) acc[i][j] = (v8f){};

  auto stage = [&](int k0, int buf) {
    int r = tid >> 1, c = (tid & 1) * 16;  // 128 rows x 32 cols, 16 halfs/thread
    const _Float16* ga = A + (long)(m0 + r) * lda + k0 + c;
    async_cp16(&sA[buf][r][c], ga);
    async_cp16(&sA[buf][r][c + 8], ga + 8);
    const _Float16* gb = Bt + (long)(n0 + r) * ldb + k0 + c;
    async_cp16(&sB[buf][r][c], gb);
    async_cp16(&sB[buf][r][c + 8], gb + 8);
  };

  const int nk = K / 32;
  stage(0, 0);
  for (int i = 0; i < nk; ++i) {
    const int cur = i & 1;
    ASYNC_WAIT_ALL();            // tile i resident (tile i+1 not yet issued)
    __syncthreads();
    if (i + 1 < nk) stage((i + 1) * 32, cur ^ 1);  // prefetch overlaps compute
    v16h a0 = frag_ld(&sA[cur][wm + lanelo][0], half);
    v16h a1 = frag_ld(&sA[cur][wm + 16 + lanelo][0], half);
#pragma unroll
    for (int j = 0; j < 4; ++j) {
      v16h bf = frag_ld(&sB[cur][wn + j * 16 + lanelo][0], half);
      acc[0][j] = wmma16(a0, bf, acc[0][j]);
      acc[1][j] = wmma16(a1, bf, acc[1][j]);
    }
    __syncthreads();             // buf cur free before it is restaged at i+2
  }

  // C/D layout: VGPR r -> M=r (lanes 0-15) / M=r+8 (lanes 16-31); N=lane%16
#pragma unroll
  for (int mi = 0; mi < 2; ++mi)
#pragma unroll
    for (int j = 0; j < 4; ++j)
#pragma unroll
      for (int r = 0; r < 8; ++r) {
        int gm = m0 + wm + mi * 16 + r + 8 * half;
        int gn = n0 + wn + j * 16 + lanelo;
        if (F16OUT) {
          ((_Float16*)Cout)[(long)gm * ldc + gn] = (_Float16)acc[mi][j][r];
        } else {
          ((float*)Cout)[(long)gm * ldc + gn] = acc[mi][j][r] + bias[gn];
        }
      }
}

// ---------------------------------------------------------------------------
// Flash attention: one WG per (b,h, 64-row Q tile). 256 threads = 8 waves.
// qkv: f16 [b,t,3072] (Q at h*64, K at 1024+h*64); vt: f16 [bh][dh][t].
// out: f16 [b,t,1024]. No scaling / no mask (faithful to reference).
// ---------------------------------------------------------------------------
__global__ __launch_bounds__(256) void attn64(const _Float16* __restrict__ qkv,
                                              const _Float16* __restrict__ vt,
                                              _Float16* __restrict__ out) {
  __shared__ _Float16 sQ[64][72];   // [m][dh]
  __shared__ _Float16 sK[64][72];   // [key][dh]
  __shared__ _Float16 sVt[64][72];  // [dh][key]
  __shared__ float    sS[64][68];
  __shared__ _Float16 sP[64][72];
  __shared__ float sMx[64], sLs[64], sAl[64];

  const int i0 = blockIdx.x * 64;
  const int bh = blockIdx.y;
  const int b = bh >> 4, h = bh & 15;
  const int tid = threadIdx.x;
  const int w = tid >> 5, lane = tid & 31;
  const int half = lane >> 4, lanelo = lane & 15;
  const int mi = w >> 1;
  const int nb = (w & 1) * 2;
  const int row = tid >> 2, col = (tid & 3) * 16;

  { // stage Q tile
    const _Float16* src = qkv + ((long)(b * 2048 + i0 + row)) * 3072 + h * 64 + col;
    async_cp16(&sQ[row][col], src);
    async_cp16(&sQ[row][col + 8], src + 8);
  }
  if (tid < 64) { sMx[tid] = -1e30f; sLs[tid] = 0.f; }

  v8f o0 = {}; v8f o1 = {};

  for (int j0 = 0; j0 < 2048; j0 += 64) {
    { // stage K tile + pre-transposed V tile (pure contiguous copies)
      const _Float16* sk =
          qkv + ((long)(b * 2048 + j0 + row)) * 3072 + 1024 + h * 64 + col;
      async_cp16(&sK[row][col], sk);
      async_cp16(&sK[row][col + 8], sk + 8);
      const _Float16* sv = vt + ((long)(bh * 64 + row)) * 2048 + j0 + col;
      async_cp16(&sVt[row][col], sv);
      async_cp16(&sVt[row][col + 8], sv + 8);
    }
    ASYNC_WAIT_ALL();
    __syncthreads();

    // S = Q * K^T (dh = 64 -> two k-steps)
    v8f s0 = {}; v8f s1 = {};
#pragma unroll
    for (int ks = 0; ks < 64; ks += 32) {
      v16h af = frag_ld(&sQ[mi * 16 + lanelo][ks], half);
      v16h b0 = frag_ld(&sK[nb * 16 + lanelo][ks], half);
      v16h b1 = frag_ld(&sK[(nb + 1) * 16 + lanelo][ks], half);
      s0 = wmma16(af, b0, s0);
      s1 = wmma16(af, b1, s1);
    }
#pragma unroll
    for (int r = 0; r < 8; ++r) {
      int m = mi * 16 + r + 8 * half;
      sS[m][nb * 16 + lanelo] = s0[r];
      sS[m][(nb + 1) * 16 + lanelo] = s1[r];
    }
    __syncthreads();

    // online softmax: one owner thread per row, float4-vectorized
    if (tid < 64) {
      const float* rp = &sS[tid][0];
      float mOld = sMx[tid];
      float4 m4 = *(const float4*)rp;
#pragma unroll
      for (int c = 1; c < 16; ++c) {
        float4 v = *(const float4*)(rp + c * 4);
        m4.x = fmaxf(m4.x, v.x); m4.y = fmaxf(m4.y, v.y);
        m4.z = fmaxf(m4.z, v.z); m4.w = fmaxf(m4.w, v.w);
      }
      float mx = fmaxf(fmaxf(fmaxf(m4.x, m4.y), fmaxf(m4.z, m4.w)), mOld);
      float al = __expf(mOld - mx);
      float a0 = 0.f, a1 = 0.f, a2 = 0.f, a3 = 0.f;
#pragma unroll
      for (int c = 0; c < 16; ++c) {
        float4 v = *(const float4*)(rp + c * 4);
        float p0 = __expf(v.x - mx), p1 = __expf(v.y - mx);
        float p2 = __expf(v.z - mx), p3 = __expf(v.w - mx);
        a0 += p0; a1 += p1; a2 += p2; a3 += p3;
        _Float16* pp = &sP[tid][c * 4];
        pp[0] = (_Float16)p0; pp[1] = (_Float16)p1;
        pp[2] = (_Float16)p2; pp[3] = (_Float16)p3;
      }
      sMx[tid] = mx;
      sLs[tid] = sLs[tid] * al + (a0 + a1) + (a2 + a3);
      sAl[tid] = al;
    }
    __syncthreads();

    // O = O*alpha + P * V
#pragma unroll
    for (int r = 0; r < 8; ++r) {
      float al = sAl[mi * 16 + r + 8 * half];
      o0[r] *= al;
      o1[r] *= al;
    }
#pragma unroll
    for (int ks = 0; ks < 64; ks += 32) {
      v16h af = frag_ld(&sP[mi * 16 + lanelo][ks], half);
      v16h b0 = frag_ld(&sVt[nb * 16 + lanelo][ks], half);
      v16h b1 = frag_ld(&sVt[(nb + 1) * 16 + lanelo][ks], half);
      o0 = wmma16(af, b0, o0);
      o1 = wmma16(af, b1, o1);
    }
    __syncthreads();
  }

  // normalize + store f16 [b, t, h*64 + d]
#pragma unroll
  for (int r = 0; r < 8; ++r) {
    int m = mi * 16 + r + 8 * half;
    float inv = 1.f / sLs[m];
    long base = ((long)(b * 2048 + i0 + m)) * 1024 + h * 64;
    out[base + nb * 16 + lanelo] = (_Float16)(o0[r] * inv);
    out[base + (nb + 1) * 16 + lanelo] = (_Float16)(o1[r] * inv);
  }
}

// ---------------------------------------------------------------------------
// launch
// ---------------------------------------------------------------------------
extern "C" void kernel_launch(void* const* d_in, const int* in_sizes, int n_in,
                              void* d_out, int out_size, void* d_ws, size_t ws_size,
                              hipStream_t stream) {
  (void)in_sizes; (void)n_in; (void)out_size; (void)ws_size;
  const float* x    = (const float*)d_in[0];  // [2,2048,1024]
  const float* wqkv = (const float*)d_in[1];  // [1024,3072]
  const float* wout = (const float*)d_in[2];  // [1024,1024]
  const float* bout = (const float*)d_in[3];  // [1024]

  char* ws = (char*)d_ws;
  _Float16* x16   = (_Float16*)(ws);                         //  8 MB: 4096x1024
  _Float16* wqT   = (_Float16*)(ws + 8ull  * 1024 * 1024);   //  6 MB: 3072x1024 (W^T)
  _Float16* woT   = (_Float16*)(ws + 14ull * 1024 * 1024);   //  2 MB: 1024x1024 (W^T)
  _Float16* qkv16 = (_Float16*)(ws + 16ull * 1024 * 1024);   // 24 MB: 4096x3072
  _Float16* vt16  = (_Float16*)(ws + 40ull * 1024 * 1024);   //  8 MB: [32][64][2048]
  _Float16* at16  = (_Float16*)(ws + 48ull * 1024 * 1024);   //  8 MB: 4096x1024

  cvt_f32_f16<<<4096, 256, 0, stream>>>(x, x16, 4194304);
  cvtT<<<dim3(96, 32), 256, 0, stream>>>(wqkv, wqT, 3072, 1024);
  cvtT<<<dim3(32, 32), 256, 0, stream>>>(wout, woT, 1024, 1024);

  // QKV projection: [4096,1024] x [1024,3072] -> f16 [4096,3072]
  gemm128<true><<<dim3(24, 32), 256, 0, stream>>>(x16, 1024, wqT, 1024,
                                                  (void*)qkv16, 3072, nullptr, 1024);
  vtrans<<<dim3(32, 32), 256, 0, stream>>>(qkv16, vt16);
  attn64<<<dim3(32, 32), 256, 0, stream>>>(qkv16, vt16, at16);

  // output projection + bias -> f32 d_out
  gemm128<false><<<dim3(8, 32), 256, 0, stream>>>(at16, 1024, woT, 1024,
                                                  d_out, 1024, bout, 1024);
}
